// TokenLearnerModule_17806934409508
// MI455X (gfx1250) — compile-verified, matrix operand order checked by reading, not run on hardware
//
#include <hip/hip_runtime.h>
#include <hip/hip_bf16.h>
#include <math.h>

// CDNA5 wave32 WMMA vector types
typedef __attribute__((ext_vector_type(16))) __bf16 v16bf;
typedef __attribute__((ext_vector_type(8)))  float  v8f;

#define LN_EPS 1e-3f
#define B_  32
#define S_  4096
#define C_  768
#define D_  64
#define T_  4

// Branch-free erf (Abramowitz & Stegun 7.1.26, max abs err ~1.5e-7).
// Straight-line VALU: rcp + 5-term Horner + v_exp_f32; no saveexec regions.
__device__ __forceinline__ float erf_fast(float v) {
  const float ax = fabsf(v);
  const float t  = __builtin_amdgcn_rcpf(1.0f + 0.3275911f * ax);
  float p = 1.061405429f;
  p = p * t - 1.453152027f;
  p = p * t + 1.421413741f;
  p = p * t - 0.284496736f;
  p = p * t + 0.254829592f;
  const float r = 1.0f - (p * t) * __expf(-ax * ax);
  return copysignf(r, v);
}

__device__ __forceinline__ float gelu_exact(float h) {
  return 0.5f * h * (1.0f + erf_fast(h * 0.70710678118654752f));
}

// ---------------------------------------------------------------------------
// Kernel 0: pack W1 [768,64] f32 -> bf16, pre-swizzled into per-lane WMMA
// B-fragment order. Fragment f = kc*4+nb. Within a fragment, lane L stores
// 16 bf16: element e -> K = kc*32 + (e/8)*16 + (L/16)*8 + (e%8),
// N = nb*16 + (L%16), matching the ISA 16-bit 32x16 B layout.
// ---------------------------------------------------------------------------
__global__ __launch_bounds__(256) void tl_pack_w1(const float* __restrict__ W1,
                                                  __bf16* __restrict__ W1p) {
  int idx = blockIdx.x * 256 + threadIdx.x;  // < 49152 == 768*64
  if (idx >= C_ * D_) return;
  int e    = idx & 15;
  int lane = (idx >> 4) & 31;
  int frag = idx >> 9;          // kc*4 + nb
  int nb   = frag & 3;
  int kc   = frag >> 2;
  int half = lane >> 4;
  int n    = lane & 15;
  int klocal = ((e >> 3) << 4) + half * 8 + (e & 7);
  int k = kc * 32 + klocal;
  W1p[idx] = (__bf16)W1[k * D_ + nb * 16 + n];
}

// ---------------------------------------------------------------------------
// Kernel 1: fused LayerNorm + (LN(x)@W1+b1) via bf16 WMMA + GELU + (@W2+b2).
// One wave32 handles TWO 16-row tiles (32 rows) so each W1 B-fragment load
// feeds two v_wmma ops (8 WMMA per K-chunk iteration).
// ---------------------------------------------------------------------------
__global__ __launch_bounds__(256) void tl_ln_mlp(
    const float* __restrict__ x, const float* __restrict__ gamma,
    const float* __restrict__ beta, const float* __restrict__ b1,
    const float* __restrict__ W2, const float* __restrict__ b2,
    const __bf16* __restrict__ W1p, float* __restrict__ logits) {
  const int  wave = threadIdx.x >> 5;
  const int  lane = threadIdx.x & 31;
  const long pair = (long)blockIdx.x * 8 + wave;   // 4096 row-pairs total
  const long row0 = pair * 32;
  const int  m16  = lane & 15;   // A-matrix row owned by this lane
  const int  half = lane >> 4;

  const float* __restrict__ xr[2] = {x + (row0 + m16) * C_,
                                     x + (row0 + 16 + m16) * C_};

  // ---- LayerNorm stats: 2 lanes per row, each covers 384 columns ----
  float mu[2], rstd[2];
#pragma unroll
  for (int u = 0; u < 2; ++u) {
    float s = 0.f, ss = 0.f;
    const float4* xv = (const float4*)(xr[u] + half * 384);
#pragma unroll 4
    for (int i = 0; i < 96; ++i) {
      float4 v = xv[i];
      s  += v.x + v.y + v.z + v.w;
      ss += v.x * v.x + v.y * v.y + v.z * v.z + v.w * v.w;
    }
    s  += __shfl_xor(s, 16, 32);
    ss += __shfl_xor(ss, 16, 32);
    mu[u]   = s * (1.f / (float)C_);
    float v = ss * (1.f / (float)C_) - mu[u] * mu[u];
    rstd[u] = rsqrtf(v + LN_EPS);
  }

  v8f acc[2][4] = {};

  for (int kc = 0; kc < 24; ++kc) {
    // Build both A fragments in ISA layout: lane m16 = M; element e -> K =
    // kc*32 + (e/8)*16 + half*8 + (e%8). Two runs of 8 consecutive floats.
    v16bf a[2];
#pragma unroll
    for (int g = 0; g < 2; ++g) {
      const int kb = kc * 32 + g * 16 + half * 8;
      float4 g0 = *(const float4*)(gamma + kb);
      float4 g1 = *(const float4*)(gamma + kb + 4);
      float4 e0 = *(const float4*)(beta + kb);
      float4 e1 = *(const float4*)(beta + kb + 4);
#pragma unroll
      for (int u = 0; u < 2; ++u) {
        float4 v0 = *(const float4*)(xr[u] + kb);
        float4 v1 = *(const float4*)(xr[u] + kb + 4);
        a[u][g * 8 + 0] = (__bf16)((v0.x - mu[u]) * rstd[u] * g0.x + e0.x);
        a[u][g * 8 + 1] = (__bf16)((v0.y - mu[u]) * rstd[u] * g0.y + e0.y);
        a[u][g * 8 + 2] = (__bf16)((v0.z - mu[u]) * rstd[u] * g0.z + e0.z);
        a[u][g * 8 + 3] = (__bf16)((v0.w - mu[u]) * rstd[u] * g0.w + e0.w);
        a[u][g * 8 + 4] = (__bf16)((v1.x - mu[u]) * rstd[u] * g1.x + e1.x);
        a[u][g * 8 + 5] = (__bf16)((v1.y - mu[u]) * rstd[u] * g1.y + e1.y);
        a[u][g * 8 + 6] = (__bf16)((v1.z - mu[u]) * rstd[u] * g1.z + e1.z);
        a[u][g * 8 + 7] = (__bf16)((v1.w - mu[u]) * rstd[u] * g1.w + e1.w);
      }
    }
    // B fragments: pre-swizzled bf16 W1 (L2-resident); each feeds 2 WMMAs.
    const v16bf* wbase = ((const v16bf*)W1p) + (size_t)kc * 128 + lane;
#pragma unroll
    for (int nb = 0; nb < 4; ++nb) {
      const v16bf bf = wbase[nb * 32];
      acc[0][nb] = __builtin_amdgcn_wmma_f32_16x16x32_bf16(
          false, a[0], false, bf, (short)0, acc[0][nb], false, false);
      acc[1][nb] = __builtin_amdgcn_wmma_f32_16x16x32_bf16(
          false, a[1], false, bf, (short)0, acc[1][nb], false, false);
    }
  }

  // ---- Epilogue. D-tile layout: VGPR j, lanes 0-15 -> (M=j, N=lane),
  // lanes 16-31 -> (M=8+j, N=lane-16). Lane holds h[m, d] for
  // m = 8*half + j and d = nb*16 + m16.
  float b1v[4], w2v[4][4];
#pragma unroll
  for (int nb = 0; nb < 4; ++nb) {
    int d = nb * 16 + m16;
    b1v[nb] = b1[d];
#pragma unroll
    for (int t = 0; t < 4; ++t) w2v[nb][t] = W2[d * T_ + t];
  }
  const float b2v0 = b2[0], b2v1 = b2[1], b2v2 = b2[2], b2v3 = b2[3];

#pragma unroll
  for (int u = 0; u < 2; ++u) {
#pragma unroll
    for (int j = 0; j < 8; ++j) {
      float p0 = 0.f, p1 = 0.f, p2 = 0.f, p3 = 0.f;
#pragma unroll
      for (int nb = 0; nb < 4; ++nb) {
        float ge = gelu_exact(acc[u][nb][j] + b1v[nb]);
        p0 += ge * w2v[nb][0];
        p1 += ge * w2v[nb][1];
        p2 += ge * w2v[nb][2];
        p3 += ge * w2v[nb][3];
      }
      // reduce over the 16 lanes of each half-wave (d dimension)
#pragma unroll
      for (int msk = 1; msk <= 8; msk <<= 1) {
        p0 += __shfl_xor(p0, msk, 32);
        p1 += __shfl_xor(p1, msk, 32);
        p2 += __shfl_xor(p2, msk, 32);
        p3 += __shfl_xor(p3, msk, 32);
      }
      if (m16 == 0) {
        const long m = row0 + u * 16 + half * 8 + j;
        float* lp = logits + m * T_;
        lp[0] = p0 + b2v0;
        lp[1] = p1 + b2v1;
        lp[2] = p2 + b2v2;
        lp[3] = p3 + b2v3;
      }
    }
  }
}

// ---------------------------------------------------------------------------
// Kernel 2: softmax over S per (b,t). 1 block per (b,t), 256 threads.
// ---------------------------------------------------------------------------
__global__ __launch_bounds__(256) void tl_softmax(const float* __restrict__ logits,
                                                  float* __restrict__ attn) {
  const int b = blockIdx.x >> 2;
  const int t = blockIdx.x & 3;
  const int tid  = threadIdx.x;
  const int lane = tid & 31, wave = tid >> 5;
  const float* lp = logits + (size_t)b * S_ * T_ + t;
  __shared__ float red[8];

  float m = -3.402823466e38f;
  for (int s = tid; s < S_; s += 256) m = fmaxf(m, lp[(size_t)s * T_]);
#pragma unroll
  for (int msk = 1; msk < 32; msk <<= 1) m = fmaxf(m, __shfl_xor(m, msk, 32));
  if (lane == 0) red[wave] = m;
  __syncthreads();
  float bm = red[0];
#pragma unroll
  for (int i = 1; i < 8; ++i) bm = fmaxf(bm, red[i]);
  __syncthreads();

  float sum = 0.f;
  for (int s = tid; s < S_; s += 256) sum += expf(lp[(size_t)s * T_] - bm);
#pragma unroll
  for (int msk = 1; msk < 32; msk <<= 1) sum += __shfl_xor(sum, msk, 32);
  if (lane == 0) red[wave] = sum;
  __syncthreads();
  float tot = 0.f;
#pragma unroll
  for (int i = 0; i < 8; ++i) tot += red[i];
  const float inv = 1.f / tot;

  float* ap = attn + ((size_t)b * T_ + t) * S_;
  for (int s = tid; s < S_; s += 256) ap[s] = expf(lp[(size_t)s * T_] - bm) * inv;
}

// ---------------------------------------------------------------------------
// Kernel 3: partial weighted sums: part[b][sl][t][c] over 512-row S slices.
// grid = 32 b * 3 c-chunks * 8 slices = 768 blocks; thread owns one column c.
// ---------------------------------------------------------------------------
__global__ __launch_bounds__(256) void tl_reduce_x(const float* __restrict__ x,
                                                   const float* __restrict__ attn,
                                                   float* __restrict__ part) {
  const int blk = blockIdx.x;
  const int b   = blk / 24;
  const int rem = blk % 24;
  const int cc  = rem / 8;
  const int sl  = rem % 8;
  const int c   = cc * 256 + threadIdx.x;

  const float* xp = x + ((size_t)b * S_ + (size_t)sl * 512) * C_ + c;
  const float* ap = attn + (size_t)b * T_ * S_ + (size_t)sl * 512;

  float a0 = 0.f, a1 = 0.f, a2 = 0.f, a3 = 0.f;
  for (int s = 0; s < 512; ++s) {
    float xvv = xp[(size_t)s * C_];
    a0 += ap[s] * xvv;
    a1 += ap[S_ + s] * xvv;
    a2 += ap[2 * S_ + s] * xvv;
    a3 += ap[3 * S_ + s] * xvv;
  }
  float* pp = part + ((size_t)(b * 8 + sl) * T_) * C_ + c;
  pp[0]      = a0;
  pp[C_]     = a1;
  pp[2 * C_] = a2;
  pp[3 * C_] = a3;
}

// Kernel 4: reduce the 8 S-slices into out[b][t][c].
__global__ __launch_bounds__(256) void tl_finalize(const float* __restrict__ part,
                                                   float* __restrict__ out) {
  const int idx = blockIdx.x * 256 + threadIdx.x;  // < 98304
  if (idx >= B_ * T_ * C_) return;
  const int b = idx / (T_ * C_);
  const int r = idx % (T_ * C_);
  const int t = r / C_;
  const int c = r % C_;
  float sum = 0.f;
#pragma unroll
  for (int sl = 0; sl < 8; ++sl)
    sum += part[((size_t)(b * 8 + sl) * T_ + t) * C_ + c];
  out[idx] = sum;
}

// ---------------------------------------------------------------------------
extern "C" void kernel_launch(void* const* d_in, const int* in_sizes, int n_in,
                              void* d_out, int out_size, void* d_ws,
                              size_t ws_size, hipStream_t stream) {
  (void)in_sizes; (void)n_in; (void)out_size; (void)ws_size;
  const float* x     = (const float*)d_in[0];
  const float* gamma = (const float*)d_in[1];
  const float* beta  = (const float*)d_in[2];
  const float* W1    = (const float*)d_in[3];
  const float* b1    = (const float*)d_in[4];
  const float* W2    = (const float*)d_in[5];
  const float* b2    = (const float*)d_in[6];
  float* out = (float*)d_out;

  // workspace layout (≈7.2 MB total)
  char*   ws     = (char*)d_ws;
  __bf16* W1p    = (__bf16*)(ws);                              // 96 KB packed W1
  float*  logits = (float*)(ws + (128 << 10));                 // 2 MB [B*S,4]
  float*  attn   = (float*)(ws + (128 << 10) + (2 << 20));     // 2 MB [B,4,S]
  float*  part   = (float*)(ws + (128 << 10) + (4 << 20));     // 3 MB partials

  tl_pack_w1 <<<192, 256, 0, stream>>>(W1, W1p);
  tl_ln_mlp  <<<512, 256, 0, stream>>>(x, gamma, beta, b1, W2, b2, W1p, logits);
  tl_softmax <<<128, 256, 0, stream>>>(logits, attn);
  tl_reduce_x<<<768, 256, 0, stream>>>(x, attn, part);
  tl_finalize<<<384, 256, 0, stream>>>(part, out);
}
